// FiBiNet_80814104641783
// MI455X (gfx1250) — matrix operand order repaired and checked by reading, not compiled, not run on hardware
//
#include <hip/hip_runtime.h>
#include <hip/hip_bf16.h>

// ---------------------------------------------------------------------------
// FiBiNet forward for MI455X (gfx1250, wave32, WMMA bf16 16x16x32)
// B=4096, F=39, E=16, pairs=741, x_con width = 2*741*16 = 23712
// MLP: 23712 -> 400 -> 400 -> 400 -> 1 with batch-norm+relu
// ---------------------------------------------------------------------------

typedef __attribute__((ext_vector_type(8)))  __bf16 bf16x8;
typedef __attribute__((ext_vector_type(16))) __bf16 bf16x16;
typedef __attribute__((ext_vector_type(8)))  float  f32x8;
typedef __attribute__((ext_vector_type(4)))  int    v4i;

#define NB 4096
#define NF 39
#define NE 16
#define NPAIR 741
#define XCW 23712      // 2*741*16
#define HID 400
#define KPAD 416       // 400 padded to multiple of 32
#define BCS 40         // LDS column stride (elements) for B tile: 80B, breaks bank conflicts

#if defined(__has_builtin)
#  if __has_builtin(__builtin_amdgcn_global_load_async_to_lds_b128)
#    define HAVE_ASYNC_LDS 1
#  endif
#endif
#ifndef HAVE_ASYNC_LDS
#  define HAVE_ASYNC_LDS 0
#endif

// -------------------------- weight conversion ------------------------------

__global__ __launch_bounds__(256) void cvt_bf16(const float* __restrict__ in,
                                                __bf16* __restrict__ out, int n) {
    int i = blockIdx.x * 256 + threadIdx.x;
    if (i < n) out[i] = (__bf16)in[i];
}

// pad K dim: in is (rows x kin) f32, out is (rows x kout) bf16, zero padded
__global__ __launch_bounds__(256) void cvt_pad_bf16(const float* __restrict__ in,
                                                    __bf16* __restrict__ out,
                                                    int rows, int kin, int kout) {
    int i = blockIdx.x * 256 + threadIdx.x;
    int n = rows * kout;
    if (i >= n) return;
    int r = i / kout, k = i % kout;
    float v = (k < kin) ? in[(size_t)r * kin + k] : 0.0f;
    out[i] = (__bf16)v;
}

// ---------------- fused gather + SENet + bilinear -> x_con (bf16) ----------

__global__ __launch_bounds__(256) void features_kernel(
    const long long* __restrict__ x, const float* __restrict__ emb,
    const float* __restrict__ w1, const float* __restrict__ b1,
    const float* __restrict__ w2, const float* __restrict__ b2,
    const float* __restrict__ Wb1, const float* __restrict__ Wb2,
    __bf16* __restrict__ XC)
{
    const int b = blockIdx.x;
    const int tid = threadIdx.x;
    __shared__ float xe[NF][NE], p1[NF][NE], p2[NF][NE];
    __shared__ float zz[NF], tt[NF], aa[NF];
    __shared__ unsigned char pi[NPAIR], pj[NPAIR];

    // gather embeddings; OFFSETS[f] = 1000*f (all field dims are 1000)
    for (int idx = tid; idx < NF * NE; idx += 256) {
        int f = idx >> 4, e = idx & 15;
        long long row = x[(size_t)b * NF + f] + (long long)f * 1000;
        xe[f][e] = emb[(size_t)row * NE + e];
    }
    // triu(39,1) pair table
    for (int p = tid; p < NPAIR; p += 256) {
        int i = 0, base = 0;
        while (base + (38 - i) <= p) { base += 38 - i; ++i; }
        pi[p] = (unsigned char)i;
        pj[p] = (unsigned char)(i + 1 + (p - base));
    }
    __syncthreads();

    if (tid < NF) {                         // z = max over E
        float m = xe[tid][0];
        #pragma unroll
        for (int e = 1; e < NE; ++e) m = fmaxf(m, xe[tid][e]);
        zz[tid] = m;
    }
    __syncthreads();
    if (tid < NF) {                         // t = relu(W1 z + b1)
        float s = b1[tid];
        for (int j = 0; j < NF; ++j) s += w1[tid * NF + j] * zz[j];
        tt[tid] = s > 0.f ? s : 0.f;
    }
    __syncthreads();
    if (tid < NF) {                         // a = relu(W2 t + b2)
        float s = b2[tid];
        for (int j = 0; j < NF; ++j) s += w2[tid * NF + j] * tt[j];
        aa[tid] = s > 0.f ? s : 0.f;
    }
    __syncthreads();
    // proj1[f] = Wb1 @ xe[f] ; proj2[f] = a[f] * (Wb2 @ xe[f])
    for (int idx = tid; idx < NF * NE; idx += 256) {
        int f = idx >> 4, d = idx & 15;
        float s1 = 0.f, s2 = 0.f;
        #pragma unroll
        for (int e = 0; e < NE; ++e) {
            float v = xe[f][e];
            s1 += Wb1[d * NE + e] * v;
            s2 += Wb2[d * NE + e] * v;
        }
        p1[f][d] = s1;
        p2[f][d] = aa[f] * s2;
    }
    __syncthreads();

    __bf16* o1 = XC + (size_t)b * XCW;      // bilinear(xe, Wb1) half
    __bf16* o2 = o1 + NPAIR * NE;           // bilinear(x_sen, Wb2) half
    for (int idx = tid; idx < NPAIR * NE; idx += 256) {
        int p = idx >> 4, e = idx & 15;
        int i = pi[p], j = pj[p];
        float vj = xe[j][e];
        o1[idx] = (__bf16)(p1[i][e] * vj);
        o2[idx] = (__bf16)(p2[i][e] * (aa[j] * vj));
    }
}

// -------------------- WMMA bf16 GEMM: C = A * W^T + bias -------------------
// A: (4096 x lda) bf16 row-major, W: (400 x ldb) bf16 row-major (N x K)
// Block = 8 waves sharing one 80-column n-tile; each wave owns a 16-row
// m-tile. Per k-step the 80x32 B panel (5KB) is staged in LDS (double
// buffered) via GLOBAL_LOAD_ASYNC_TO_LDS (ASYNCcnt) when available, so all
// 8 waves read B from LDS instead of 8x redundant L2 traffic.
// Grid: (4096/128 m-groups) x 5 n-tiles = 160 blocks.

__global__ __launch_bounds__(256) void gemm_wmma_bf16(
    const __bf16* __restrict__ A, int lda,
    const __bf16* __restrict__ Bw, int ldb,
    const float* __restrict__ bias,
    float* __restrict__ C, int ldc, int ksteps)
{
    const int lane = threadIdx.x & 31;
    const int wave = threadIdx.x >> 5;
    const int n0 = (blockIdx.x % 5) * 80;
    const int m0 = (blockIdx.x / 5) * 128 + wave * 16;
    const int grp = lane >> 4;              // 0: lanes 0-15, 1: lanes 16-31
    const int l15 = lane & 15;

    // B panel: 80 columns x 32 K, column stride BCS=40 elems (80B) anti-conflict pad
    __shared__ __align__(128) __bf16 bsh[2][80 * BCS];

    f32x8 acc[5];
    #pragma unroll
    for (int s = 0; s < 5; ++s) { f32x8 z = {}; acc[s] = z; }

    const __bf16* arow = A + (size_t)(m0 + l15) * lda;

#if HAVE_ASYNC_LDS
    // 320 x 16B chunks: chunk t -> column t/4, 8-element part t%4
    auto stage = [&](int kt, int buf) {
        const int kk = kt * 32;
        for (int t = threadIdx.x; t < 320; t += 256) {
            const int col = t >> 2, part = t & 3;
            const __bf16* g = Bw + (size_t)(n0 + col) * ldb + kk + part * 8;
            __builtin_amdgcn_global_load_async_to_lds_b128(
                (__attribute__((address_space(1))) v4i*)g,
                (__attribute__((address_space(3))) v4i*)&bsh[buf][col * BCS + part * 8],
                0, 0);
        }
    };
    stage(0, 0);
#else
    bf16x8 sreg[2];
    auto stage = [&](int kt, int) {     // global -> VGPR
        const int kk = kt * 32;
        #pragma unroll
        for (int u = 0; u < 2; ++u) {
            const int t = threadIdx.x + u * 256;
            if (t < 320) {
                const int col = t >> 2, part = t & 3;
                sreg[u] = *(const bf16x8*)(Bw + (size_t)(n0 + col) * ldb + kk + part * 8);
            }
        }
    };
    stage(0, 0);
#endif

    for (int kt = 0; kt < ksteps; ++kt) {
        const int buf = kt & 1;
#if HAVE_ASYNC_LDS
#  if __has_builtin(__builtin_amdgcn_s_wait_asynccnt)
        __builtin_amdgcn_s_wait_asynccnt(0);
#  else
        asm volatile("s_wait_asynccnt 0" ::: "memory");
#  endif
#else
        // commit staged regs -> LDS
        #pragma unroll
        for (int u = 0; u < 2; ++u) {
            const int t = threadIdx.x + u * 256;
            if (t < 320) {
                const int col = t >> 2, part = t & 3;
                *(bf16x8*)&bsh[buf][col * BCS + part * 8] = sreg[u];
            }
        }
#endif
        __syncthreads();                              // panel[buf] visible to all waves
        if (kt + 1 < ksteps) stage(kt + 1, buf ^ 1);  // overlap next panel with compute

        const int kk = kt * 32;
        // A fragment (16x32): lanes 0-15 M=lane, K {kk..+7, kk+16..+23};
        // lanes 16-31 K {kk+8..+15, kk+24..+31}
        bf16x8 a0 = *(const bf16x8*)(arow + kk + grp * 8);
        bf16x8 a1 = *(const bf16x8*)(arow + kk + 16 + grp * 8);
        bf16x16 af;
        #pragma unroll
        for (int i = 0; i < 8; ++i) { af[i] = a0[i]; af[8 + i] = a1[i]; }
        __builtin_prefetch(arow + kk + 512, 0, 3);    // near-cache prefetch on A stream

        #pragma unroll
        for (int s = 0; s < 5; ++s) {
            // B fragment (32x16): lane holds column s*16+l15, K {grp*8..+7, 16+grp*8..+7}
            const __bf16* bp = &bsh[buf][(s * 16 + l15) * BCS + grp * 8];
            bf16x8 b0 = *(const bf16x8*)bp;
            bf16x8 b1 = *(const bf16x8*)(bp + 16);
            bf16x16 bfrag;
            #pragma unroll
            for (int i = 0; i < 8; ++i) { bfrag[i] = b0[i]; bfrag[8 + i] = b1[i]; }
            acc[s] = __builtin_amdgcn_wmma_f32_16x16x32_bf16(
                false, af, false, bfrag, (short)0, acc[s], false, false);
        }
        __syncthreads();                              // readers done before panel re-staged
    }

    #pragma unroll
    for (int s = 0; s < 5; ++s) {
        const int col = n0 + s * 16 + l15;
        const float bv = bias[col];
        #pragma unroll
        for (int r = 0; r < 8; ++r) {
            const int row = m0 + r + grp * 8;   // C/D layout: vgpr r -> M=r (+8 for hi lanes)
            C[(size_t)row * ldc + col] = acc[s][r] + bv;
        }
    }
}

// ----------------------------- batch norm ----------------------------------

__global__ __launch_bounds__(256) void bn_stats(const float* __restrict__ H,
                                                float* __restrict__ mu,
                                                float* __restrict__ rs) {
    const int c = blockIdx.x;               // 0..399
    float s = 0.f, s2 = 0.f;
    for (int r = threadIdx.x; r < NB; r += 256) {
        float v = H[(size_t)r * HID + c];
        s += v; s2 += v * v;
    }
    __shared__ float sh[256], sh2[256];
    sh[threadIdx.x] = s; sh2[threadIdx.x] = s2;
    __syncthreads();
    for (int o = 128; o > 0; o >>= 1) {
        if (threadIdx.x < o) { sh[threadIdx.x] += sh[threadIdx.x + o];
                               sh2[threadIdx.x] += sh2[threadIdx.x + o]; }
        __syncthreads();
    }
    if (threadIdx.x == 0) {
        float m = sh[0] * (1.0f / NB);
        float v = sh2[0] * (1.0f / NB) - m * m;
        mu[c] = m;
        rs[c] = rsqrtf(v + 1e-5f);
    }
}

// y = relu(g*(h-mu)*rstd + be) -> bf16, K-padded to 416 with zeros
__global__ __launch_bounds__(256) void bn_apply(const float* __restrict__ H,
                                                const float* __restrict__ mu,
                                                const float* __restrict__ rs,
                                                const float* __restrict__ g,
                                                const float* __restrict__ be,
                                                __bf16* __restrict__ Aout) {
    int idx = blockIdx.x * 256 + threadIdx.x;
    if (idx >= NB * KPAD) return;
    int r = idx / KPAD, k = idx % KPAD;
    float y = 0.f;
    if (k < HID) {
        float h = H[(size_t)r * HID + k];
        y = g[k] * (h - mu[k]) * rs[k] + be[k];
        y = y > 0.f ? y : 0.f;
    }
    Aout[idx] = (__bf16)y;
}

// ------------------------- final 400 -> 1 layer ----------------------------

__global__ __launch_bounds__(256) void final_dot(const __bf16* __restrict__ A1,
                                                 const float* __restrict__ w,
                                                 const float* __restrict__ b,
                                                 float* __restrict__ out) {
    const int lane = threadIdx.x & 31;
    const int row = blockIdx.x * 8 + (threadIdx.x >> 5);
    const __bf16* ar = A1 + (size_t)row * KPAD;
    float s = 0.f;
    for (int k = lane; k < HID; k += 32) s += (float)ar[k] * w[k];
    for (int o = 16; o > 0; o >>= 1) s += __shfl_down(s, o, 32);
    if (lane == 0) out[row] = s + b[0];
}

// ------------------------------- launcher ----------------------------------

extern "C" void kernel_launch(void* const* d_in, const int* in_sizes, int n_in,
                              void* d_out, int out_size, void* d_ws, size_t ws_size,
                              hipStream_t stream) {
    const long long* x  = (const long long*)d_in[0];
    const float* emb    = (const float*)d_in[1];
    const float* sen_w1 = (const float*)d_in[2];
    const float* sen_b1 = (const float*)d_in[3];
    const float* sen_w2 = (const float*)d_in[4];
    const float* sen_b2 = (const float*)d_in[5];
    const float* Wb1    = (const float*)d_in[6];
    const float* Wb2    = (const float*)d_in[7];
    const float* mw0 = (const float*)d_in[8];   const float* mb0 = (const float*)d_in[9];
    const float* g0  = (const float*)d_in[10];  const float* be0 = (const float*)d_in[11];
    const float* mw1 = (const float*)d_in[12];  const float* mb1 = (const float*)d_in[13];
    const float* g1  = (const float*)d_in[14];  const float* be1 = (const float*)d_in[15];
    const float* mw2 = (const float*)d_in[16];  const float* mb2 = (const float*)d_in[17];
    const float* g2  = (const float*)d_in[18];  const float* be2 = (const float*)d_in[19];
    const float* mw3 = (const float*)d_in[20];  const float* mb3 = (const float*)d_in[21];
    float* out = (float*)d_out;

    // workspace carve-out (256B aligned)
    char* ws = (char*)d_ws;
    size_t off = 0;
    auto carve = [&](size_t bytes) -> char* {
        char* p = ws + off;
        off = (off + bytes + 255) & ~(size_t)255;
        return p;
    };
    __bf16* W0b = (__bf16*)carve((size_t)HID * XCW  * 2);   // 19.0 MB
    __bf16* W1b = (__bf16*)carve((size_t)HID * KPAD * 2);
    __bf16* W2b = (__bf16*)carve((size_t)HID * KPAD * 2);
    __bf16* XC  = (__bf16*)carve((size_t)NB  * XCW  * 2);   // 194 MB (~fits L2)
    float*  HP  = (float*) carve((size_t)NB  * HID  * 4);
    __bf16* A1  = (__bf16*)carve((size_t)NB  * KPAD * 2);
    __bf16* A2  = (__bf16*)carve((size_t)NB  * KPAD * 2);
    float*  MU  = (float*) carve(HID * 4);
    float*  RS  = (float*) carve(HID * 4);

    // weights -> bf16 (K padded for layers 1/2)
    {
        int n0 = HID * XCW;
        cvt_bf16<<<(n0 + 255) / 256, 256, 0, stream>>>(mw0, W0b, n0);
        int n1 = HID * KPAD;
        cvt_pad_bf16<<<(n1 + 255) / 256, 256, 0, stream>>>(mw1, W1b, HID, HID, KPAD);
        cvt_pad_bf16<<<(n1 + 255) / 256, 256, 0, stream>>>(mw2, W2b, HID, HID, KPAD);
    }

    // fused gather + SENet + bilinear -> x_con bf16
    features_kernel<<<NB, 256, 0, stream>>>(x, emb, sen_w1, sen_b1, sen_w2, sen_b2,
                                            Wb1, Wb2, XC);

    const int GEMM_BLOCKS = (NB / 128) * 5;   // 160

    // layer 0: K = 23712 (741 k-steps)
    gemm_wmma_bf16<<<GEMM_BLOCKS, 256, 0, stream>>>(XC, XCW, W0b, XCW, mb0, HP, HID, XCW / 32);
    bn_stats<<<HID, 256, 0, stream>>>(HP, MU, RS);
    bn_apply<<<(NB * KPAD + 255) / 256, 256, 0, stream>>>(HP, MU, RS, g0, be0, A1);

    // layer 1: K = 416 (13 k-steps)
    gemm_wmma_bf16<<<GEMM_BLOCKS, 256, 0, stream>>>(A1, KPAD, W1b, KPAD, mb1, HP, HID, KPAD / 32);
    bn_stats<<<HID, 256, 0, stream>>>(HP, MU, RS);
    bn_apply<<<(NB * KPAD + 255) / 256, 256, 0, stream>>>(HP, MU, RS, g1, be1, A2);

    // layer 2
    gemm_wmma_bf16<<<GEMM_BLOCKS, 256, 0, stream>>>(A2, KPAD, W2b, KPAD, mb2, HP, HID, KPAD / 32);
    bn_stats<<<HID, 256, 0, stream>>>(HP, MU, RS);
    bn_apply<<<(NB * KPAD + 255) / 256, 256, 0, stream>>>(HP, MU, RS, g2, be2, A1);

    // final 400 -> 1
    final_dot<<<NB / 8, 256, 0, stream>>>(A1, mw3, mb3, out);
}
